// GPSLayer_86285892976716
// MI455X (gfx1250) — matrix operand order repaired and checked by reading, not compiled, not run on hardware
//
#include <hip/hip_runtime.h>
#include <hip/hip_bf16.h>

// ---------------------------------------------------------------------------
// GPS layer for MI455X (gfx1250, wave32, WMMA bf16 w/ fp32 accumulate)
// ---------------------------------------------------------------------------

#define GN 8192      // nodes
#define GD 256       // feature dim
#define GH 4         // heads
#define GHD 64       // head dim
#define GE 524288    // edges

typedef __attribute__((ext_vector_type(16))) __bf16 v16bf;
typedef __attribute__((ext_vector_type(8)))  float  v8f;

__device__ __forceinline__ __bf16 bf(float x) { return (__bf16)x; }

// --- WMMA fragment loaders (CDNA5 16-bit layouts, K-major memory) ----------
// A 16x32: lane m (m=lane&15) holds K = {kc..kc+7} U {kc+16..kc+23}, kc = (lane>=16)?8:0
__device__ __forceinline__ v16bf ld_a(const __bf16* base, int ld) {
  const int lane = threadIdx.x & 31;
  const __bf16* p = base + (size_t)(lane & 15) * ld + ((lane >> 4) << 3);
  v16bf f;
  reinterpret_cast<uint4*>(&f)[0] = *reinterpret_cast<const uint4*>(p);
  reinterpret_cast<uint4*>(&f)[1] = *reinterpret_cast<const uint4*>(p + 16);
  return f;
}
// B 32x16: lane n (n=lane&15) holds K = {kc..kc+7} U {kc+16..kc+23}, kc = (lane>=16)?8:0
__device__ __forceinline__ v16bf ld_b(const __bf16* base, int ld) {
  const int lane = threadIdx.x & 31;
  const __bf16* p = base + (size_t)(lane & 15) * ld + ((lane >> 4) << 3);
  v16bf f;
  reinterpret_cast<uint4*>(&f)[0] = *reinterpret_cast<const uint4*>(p);
  reinterpret_cast<uint4*>(&f)[1] = *reinterpret_cast<const uint4*>(p + 16);
  return f;
}

#define WMMA_BF16(A, B, C) \
  __builtin_amdgcn_wmma_f32_16x16x32_bf16(false, (A), false, (B), (short)0, (C), false, false)

// --- fp32 -> bf16 conversion ------------------------------------------------
__global__ void k_cvt_bf16(const float* __restrict__ in, __bf16* __restrict__ out, int n) {
  int i = blockIdx.x * blockDim.x + threadIdx.x;
  if (i < n) out[i] = bf(in[i]);
}

// --- GEMM: C[M,Nout] = A[M,K] @ B[Nout,K]^T + bias, wave -> 32x32 tile ------
// epi bit0: ReLU, bit1: store bf16 to Cb (else fp32 to Cf)
__global__ __launch_bounds__(256) void k_gemm_bt(
    const __bf16* __restrict__ A, const __bf16* __restrict__ B,
    const float* __restrict__ bias, float* __restrict__ Cf, __bf16* __restrict__ Cb,
    int M, int Nout, int K, int epi) {
  const int wave   = (blockIdx.x * blockDim.x + threadIdx.x) >> 5;
  const int tilesN = Nout >> 5;
  const int tm = wave / tilesN, tn = wave % tilesN;
  if (tm * 32 >= M) return;
  const int lane = threadIdx.x & 31;
  v8f acc[2][2] = {};
  const __bf16* Ab = A + (size_t)tm * 32 * K;
  const __bf16* Bb = B + (size_t)tn * 32 * K;
  for (int k0 = 0; k0 < K; k0 += 32) {
    v16bf a0 = ld_a(Ab + k0, K);
    v16bf a1 = ld_a(Ab + (size_t)16 * K + k0, K);
    v16bf b0 = ld_b(Bb + k0, K);
    v16bf b1 = ld_b(Bb + (size_t)16 * K + k0, K);
    acc[0][0] = WMMA_BF16(a0, b0, acc[0][0]);
    acc[0][1] = WMMA_BF16(a0, b1, acc[0][1]);
    acc[1][0] = WMMA_BF16(a1, b0, acc[1][0]);
    acc[1][1] = WMMA_BF16(a1, b1, acc[1][1]);
  }
  const int col = lane & 15;
  const int rb  = (lane >> 4) << 3;
#pragma unroll
  for (int i = 0; i < 2; ++i)
#pragma unroll
    for (int j = 0; j < 2; ++j) {
      const int gcol = tn * 32 + j * 16 + col;
      const float bv = bias ? bias[gcol] : 0.f;
#pragma unroll
      for (int r = 0; r < 8; ++r) {
        const int grow = tm * 32 + i * 16 + rb + r;
        float v = acc[i][j][r] + bv;
        if (epi & 1) v = fmaxf(v, 0.f);
        if (epi & 2) Cb[(size_t)grow * Nout + gcol] = bf(v);
        else         Cf[(size_t)grow * Nout + gcol] = v;
      }
    }
}

// --- ResGated edge scatter: wave per edge, 8 floats/lane --------------------
__global__ __launch_bounds__(256) void k_edge(
    const long long* __restrict__ ei, const float* __restrict__ h2,
    const float* __restrict__ t3, const float* __restrict__ t4,
    float* __restrict__ ssum, float* __restrict__ cnt) {
  const int wave = (blockIdx.x * blockDim.x + threadIdx.x) >> 5;
  if (wave >= GE) return;
  const int lane = threadIdx.x & 31;
  const long long s = ei[wave];
  const long long d = ei[(size_t)GE + wave];
  const float4* h2s = (const float4*)(h2 + (size_t)s * GD);
  const float4* t3s = (const float4*)(t3 + (size_t)s * GD);
  const float4* t4d = (const float4*)(t4 + (size_t)d * GD);
  float* out = ssum + (size_t)d * GD;
#pragma unroll
  for (int c = 0; c < 2; ++c) {
    const int vi = lane * 2 + c;
    const float4 a = t3s[vi], b = t4d[vi], m = h2s[vi];
    float4 g;
    g.x = m.x / (1.f + __expf(-(a.x + b.x)));
    g.y = m.y / (1.f + __expf(-(a.y + b.y)));
    g.z = m.z / (1.f + __expf(-(a.z + b.z)));
    g.w = m.w / (1.f + __expf(-(a.w + b.w)));
    unsafeAtomicAdd(&out[vi * 4 + 0], g.x);
    unsafeAtomicAdd(&out[vi * 4 + 1], g.y);
    unsafeAtomicAdd(&out[vi * 4 + 2], g.z);
    unsafeAtomicAdd(&out[vi * 4 + 3], g.w);
  }
  if (lane == 0) unsafeAtomicAdd(&cnt[d], 1.f);
}

// --- z = (h1+ssum)/max(cnt,1) + hres ----------------------------------------
__global__ void k_local_pre(const float* __restrict__ h1, const float* __restrict__ ssum,
                            const float* __restrict__ cnt, const float* __restrict__ hres,
                            float* __restrict__ z) {
  const int i = blockIdx.x * blockDim.x + threadIdx.x;
  const int n = i >> 8;  // / GD
  z[i] = (h1[i] + ssum[i]) / fmaxf(cnt[n], 1.f) + hres[i];
}

// --- out = LayerNorm(relu(pre)) * g + b + resid : wave per row --------------
__global__ __launch_bounds__(256) void k_ln_relu_add(
    const float* __restrict__ pre, const float* __restrict__ resid,
    const float* __restrict__ g, const float* __restrict__ b, float* __restrict__ out) {
  const int row  = (blockIdx.x * blockDim.x + threadIdx.x) >> 5;
  const int lane = threadIdx.x & 31;
  const float* p = pre + (size_t)row * GD;
  float v[8], s = 0.f;
#pragma unroll
  for (int c = 0; c < 8; ++c) { float t = fmaxf(p[lane * 8 + c], 0.f); v[c] = t; s += t; }
  for (int m = 16; m > 0; m >>= 1) s += __shfl_xor(s, m);
  const float mu = s * (1.f / GD);
  float vs = 0.f;
#pragma unroll
  for (int c = 0; c < 8; ++c) { float t = v[c] - mu; vs += t * t; }
  for (int m = 16; m > 0; m >>= 1) vs += __shfl_xor(vs, m);
  const float inv = rsqrtf(vs * (1.f / GD) + 1e-5f);
#pragma unroll
  for (int c = 0; c < 8; ++c) {
    const int j = lane * 8 + c;
    out[(size_t)row * GD + j] = (v[c] - mu) * inv * g[j] + b[j] + resid[(size_t)row * GD + j];
  }
}

// --- split qkv fp32 [N,3D] into bf16 q/k [H][N][HD] and vT [H][HD][N] -------
__global__ void k_pack_qkv(const float* __restrict__ qkv, __bf16* __restrict__ qb,
                           __bf16* __restrict__ kb, __bf16* __restrict__ vt) {
  const int i = blockIdx.x * blockDim.x + threadIdx.x;  // over N*D
  const int n = i >> 8, hd = i & 255;
  const int h = hd >> 6, d = hd & 63;
  const float* row = qkv + (size_t)n * (3 * GD);
  qb[((size_t)h * GN + n) * GHD + d] = bf(row[hd]);
  kb[((size_t)h * GN + n) * GHD + d] = bf(row[GD + hd]);
  vt[((size_t)h * GHD + d) * GN + n] = bf(row[2 * GD + hd]);
}

// --- flash attention: wave per (head, 16 query rows) ------------------------
__global__ __launch_bounds__(256) void k_attn(
    const __bf16* __restrict__ qb, const __bf16* __restrict__ kb,
    const __bf16* __restrict__ vt, __bf16* __restrict__ obuf) {
  __shared__ __align__(16) __bf16 lds[8][16 * 32];   // P staging, 1KB per wave
  const int wib  = threadIdx.x >> 5;
  const int wave = (blockIdx.x * blockDim.x + threadIdx.x) >> 5;
  const int lane = threadIdx.x & 31;
  const int h  = wave >> 9;              // / (N/16)
  const int q0 = (wave & 511) * 16;
  const __bf16* qh = qb + (size_t)h * GN * GHD;
  const __bf16* kh = kb + (size_t)h * GN * GHD;
  const __bf16* vh = vt + (size_t)h * GHD * GN;
  __bf16* pl = &lds[wib][0];

  const v16bf qf0 = ld_a(qh + (size_t)q0 * GHD, GHD);
  const v16bf qf1 = ld_a(qh + (size_t)q0 * GHD + 32, GHD);
  v8f o[4] = {};
  float rm[8], rl[8];
#pragma unroll
  for (int r = 0; r < 8; ++r) { rm[r] = -1e30f; rl[r] = 0.f; }
  const int col = lane & 15;
  const int rb  = (lane >> 4) << 3;
  const float scale = 0.125f;  // 1/sqrt(64)

  for (int k0 = 0; k0 < GN; k0 += 32) {
    v8f s0 = {}, s1 = {};
    s0 = WMMA_BF16(qf0, ld_b(kh + (size_t)k0 * GHD, GHD), s0);
    s0 = WMMA_BF16(qf1, ld_b(kh + (size_t)k0 * GHD + 32, GHD), s0);
    s1 = WMMA_BF16(qf0, ld_b(kh + (size_t)(k0 + 16) * GHD, GHD), s1);
    s1 = WMMA_BF16(qf1, ld_b(kh + (size_t)(k0 + 16) * GHD + 32, GHD), s1);
    // online softmax per row (rows rb..rb+7 live in this 16-lane half)
#pragma unroll
    for (int r = 0; r < 8; ++r) {
      const float a = s0[r] * scale, bvl = s1[r] * scale;
      float mx = fmaxf(a, bvl);
      mx = fmaxf(mx, __shfl_xor(mx, 1));
      mx = fmaxf(mx, __shfl_xor(mx, 2));
      mx = fmaxf(mx, __shfl_xor(mx, 4));
      mx = fmaxf(mx, __shfl_xor(mx, 8));
      const float mn = fmaxf(rm[r], mx);
      const float cf = __expf(rm[r] - mn);
      const float p0 = __expf(a - mn), p1 = __expf(bvl - mn);
      float ps = p0 + p1;
      ps += __shfl_xor(ps, 1); ps += __shfl_xor(ps, 2);
      ps += __shfl_xor(ps, 4); ps += __shfl_xor(ps, 8);
      rl[r] = rl[r] * cf + ps;
      rm[r] = mn;
      o[0][r] *= cf; o[1][r] *= cf; o[2][r] *= cf; o[3][r] *= cf;
      const int row = rb + r;
      pl[row * 32 + col]      = bf(p0);
      pl[row * 32 + 16 + col] = bf(p1);
    }
    const v16bf pf = ld_a(pl, 32);  // C-layout -> A-frag via LDS
#pragma unroll
    for (int t = 0; t < 4; ++t) {
      const v16bf vf = ld_b(vh + (size_t)(t * 16) * GN + k0, GN);
      o[t] = WMMA_BF16(pf, vf, o[t]);
    }
  }
#pragma unroll
  for (int t = 0; t < 4; ++t)
#pragma unroll
    for (int r = 0; r < 8; ++r)
      obuf[(size_t)(q0 + rb + r) * GD + h * GHD + t * 16 + col] = bf(o[t][r] / rl[r]);
}

// --- outf = a + b; outb = bf16(outf) ----------------------------------------
__global__ void k_add_cvt(const float* __restrict__ a, const float* __restrict__ b,
                          float* __restrict__ of, __bf16* __restrict__ ob, int n) {
  const int i = blockIdx.x * blockDim.x + threadIdx.x;
  if (i < n) { const float v = a[i] + b[i]; of[i] = v; ob[i] = bf(v); }
}

// --- y = L2normalize(out + ff) : wave per row -------------------------------
__global__ __launch_bounds__(256) void k_final(const float* __restrict__ outf,
                                               const float* __restrict__ ff,
                                               float* __restrict__ y) {
  const int row  = (blockIdx.x * blockDim.x + threadIdx.x) >> 5;
  const int lane = threadIdx.x & 31;
  float v[8], s = 0.f;
#pragma unroll
  for (int c = 0; c < 8; ++c) {
    const size_t j = (size_t)row * GD + lane * 8 + c;
    const float t = outf[j] + ff[j];
    v[c] = t; s += t * t;
  }
  for (int m = 16; m > 0; m >>= 1) s += __shfl_xor(s, m);
  const float inv = 1.f / fmaxf(sqrtf(s), 1e-12f);
#pragma unroll
  for (int c = 0; c < 8; ++c) y[(size_t)row * GD + lane * 8 + c] = v[c] * inv;
}

// ---------------------------------------------------------------------------
extern "C" void kernel_launch(void* const* d_in, const int* in_sizes, int n_in,
                              void* d_out, int out_size, void* d_ws, size_t ws_size,
                              hipStream_t stream) {
  (void)in_sizes; (void)n_in; (void)out_size; (void)ws_size;
  const float* x      = (const float*)d_in[0];
  const long long* ei = (const long long*)d_in[1];
  const float* w1 = (const float*)d_in[2],  *b1 = (const float*)d_in[3];
  const float* w2 = (const float*)d_in[4],  *b2 = (const float*)d_in[5];
  const float* w3 = (const float*)d_in[6],  *b3 = (const float*)d_in[7];
  const float* w4 = (const float*)d_in[8],  *b4 = (const float*)d_in[9];
  const float* wres = (const float*)d_in[10], *bres = (const float*)d_in[11];
  const float* lng = (const float*)d_in[12], *lnb = (const float*)d_in[13];
  const float* lag = (const float*)d_in[14], *lab = (const float*)d_in[15];
  const float* ipw = (const float*)d_in[16], *ipb = (const float*)d_in[17];
  const float* opw = (const float*)d_in[18], *opb = (const float*)d_in[19];
  const float* f1w = (const float*)d_in[20], *f1b = (const float*)d_in[21];
  const float* f2w = (const float*)d_in[22], *f2b = (const float*)d_in[23];
  float* y = (float*)d_out;

  char* ws = (char*)d_ws;
  size_t off = 0;
  auto take = [&](size_t bytes) -> void* {
    void* p = ws + off; off += (bytes + 255) & ~(size_t)255; return p;
  };

  const size_t ND  = (size_t)GN * GD;       // 2,097,152
  const size_t DD  = (size_t)GD * GD;       // 65,536
  __bf16* xb   = (__bf16*)take(ND * 2);
  __bf16* w1b  = (__bf16*)take(DD * 2);
  __bf16* w2b  = (__bf16*)take(DD * 2);
  __bf16* w3b  = (__bf16*)take(DD * 2);
  __bf16* w4b  = (__bf16*)take(DD * 2);
  __bf16* wrb  = (__bf16*)take(DD * 2);
  __bf16* ipwb = (__bf16*)take(3 * DD * 2);
  __bf16* opwb = (__bf16*)take(DD * 2);
  __bf16* f1wb = (__bf16*)take(2 * DD * 2);
  __bf16* f2wb = (__bf16*)take(2 * DD * 2);
  // six 8MB fp32 scratch buffers (contiguous; aliased across phases)
  float* buf0 = (float*)take(ND * 4);
  float* buf1 = (float*)take(ND * 4);
  float* buf2 = (float*)take(ND * 4);
  float* buf3 = (float*)take(ND * 4);
  float* buf4 = (float*)take(ND * 4);
  float* buf5 = (float*)take(ND * 4);
  float* cnt    = (float*)take((size_t)GN * 4);
  float* zloc   = (float*)take(ND * 4);
  float* localf = (float*)take(ND * 4);

  auto cvt = [&](const float* src, __bf16* dst, int n) {
    k_cvt_bf16<<<(n + 255) / 256, 256, 0, stream>>>(src, dst, n);
  };
  auto gemm = [&](const __bf16* A, const __bf16* B, const float* bias,
                  float* Cf, __bf16* Cb, int M, int Nout, int K, int epi) {
    const int waves = (M / 32) * (Nout / 32);
    k_gemm_bt<<<waves * 32 / 256, 256, 0, stream>>>(A, B, bias, Cf, Cb, M, Nout, K, epi);
  };

  // 0) fp32 -> bf16 staging
  cvt(x, xb, (int)ND);
  cvt(w1, w1b, (int)DD);   cvt(w2, w2b, (int)DD);
  cvt(w3, w3b, (int)DD);   cvt(w4, w4b, (int)DD);
  cvt(wres, wrb, (int)DD); cvt(ipw, ipwb, (int)(3 * DD));
  cvt(opw, opwb, (int)DD); cvt(f1w, f1wb, (int)(2 * DD));
  cvt(f2w, f2wb, (int)(2 * DD));

  // 1) local-path projections (WMMA GEMMs)
  gemm(xb, w1b, b1,   buf0, nullptr, GN, GD, GD, 0);  // h1
  gemm(xb, w2b, b2,   buf1, nullptr, GN, GD, GD, 0);  // h2
  gemm(xb, w3b, b3,   buf2, nullptr, GN, GD, GD, 0);  // t3
  gemm(xb, w4b, b4,   buf3, nullptr, GN, GD, GD, 0);  // t4
  gemm(xb, wrb, bres, buf4, nullptr, GN, GD, GD, 0);  // hres

  // 2) gated message passing (bandwidth-bound: ~2.7GB -> ~120us @ 23.3TB/s)
  hipMemsetAsync(buf5, 0, ND * 4, stream);            // ssum
  hipMemsetAsync(cnt,  0, (size_t)GN * 4, stream);
  k_edge<<<(GE * 32) / 256, 256, 0, stream>>>(ei, buf1, buf2, buf3, buf5, cnt);
  k_local_pre<<<(int)(ND / 256), 256, 0, stream>>>(buf0, buf5, cnt, buf4, zloc);
  k_ln_relu_add<<<GN * 32 / 256, 256, 0, stream>>>(zloc, x, lng, lnb, localf);

  // 3) attention (dominant ~6.9e10 FLOPs -> all WMMA)
  float* qkvf = buf0;                                  // 24MB spans buf0..buf2
  gemm(xb, ipwb, ipb, qkvf, nullptr, GN, 3 * GD, GD, 0);
  __bf16* qbuf = (__bf16*)buf3;
  __bf16* kbuf = (__bf16*)buf4;
  __bf16* vtb  = (__bf16*)buf5;
  k_pack_qkv<<<(int)(ND / 256), 256, 0, stream>>>(qkvf, qbuf, kbuf, vtb);
  __bf16* obuf = (__bf16*)zloc;
  k_attn<<<(GH * (GN / 16)) * 32 / 256, 256, 0, stream>>>(qbuf, kbuf, vtb, obuf);
  float* attnproj = buf0;
  gemm(obuf, opwb, opb, attnproj, nullptr, GN, GD, GD, 0);
  float* attnf = buf1;
  k_ln_relu_add<<<GN * 32 / 256, 256, 0, stream>>>(attnproj, x, lag, lab, attnf);

  // 4) FFN + final L2 normalize
  float*  outf = buf2;
  __bf16* outb = (__bf16*)buf3;
  k_add_cvt<<<(int)(ND / 256), 256, 0, stream>>>(localf, attnf, outf, outb, (int)ND);
  __bf16* ffh = (__bf16*)buf4;                         // [N, 512] bf16 = 8MB
  gemm(outb, f1wb, f1b, nullptr, ffh, GN, 2 * GD, GD, 3);        // relu + bf16
  float* ffout = buf5;
  gemm(ffh, f2wb, f2b, ffout, nullptr, GN, GD, 2 * GD, 0);
  k_final<<<GN * 32 / 256, 256, 0, stream>>>(outf, ffout, y);
}
